// FeedbackDetector_8589934869
// MI455X (gfx1250) — matrix operand order repaired and checked by reading, not compiled, not run on hardware
//
#include <hip/hip_runtime.h>
#include <hip/hip_bf16.h>

// ---------------- problem constants ----------------
#define BB      16
#define NFREQ   481
#define TFRAMES 500
#define NBINS   (BB * NFREQ)          // 7696
#define FCH     16
#define HID     32

typedef __attribute__((ext_vector_type(16))) __bf16 v16bf;
typedef __attribute__((ext_vector_type(8)))  __bf16 v8bf;
typedef __attribute__((ext_vector_type(8)))  float  v8f;

__device__ __forceinline__ float sigm_f(float x) {
    return __builtin_amdgcn_rcpf(1.0f + __expf(-x));
}
__device__ __forceinline__ float tanh_f(float x) {
    float e = __expf(-2.0f * x);
    return (1.0f - e) * __builtin_amdgcn_rcpf(1.0f + e);
}

// =====================================================================
// Stage 1: per-frame conv over frequency (conv1 k=9 pad4 -> PReLU ->
//          conv2 k=5 pad2 -> PReLU), output bf16 in (t, n, c) layout.
// One block = one (b,t) frame; 256 threads cover 481 freq bins.
// =====================================================================
__global__ __launch_bounds__(256) void fbd_conv_kernel(
    const float* __restrict__ feat,   // (B, 4, F, T)
    const float* __restrict__ w1,     // (16, 4, 9)
    const float* __restrict__ b1,     // (16)
    const float* __restrict__ a1p,    // (1)
    const float* __restrict__ w2,     // (16, 16, 5)
    const float* __restrict__ b2,     // (16)
    const float* __restrict__ a2p,    // (1)
    __bf16* __restrict__ xout)        // (T, NBINS, 16) bf16
{
    __shared__ float x1[16 * 485];    // conv1 output, padded by 2 each side

    const int bt  = blockIdx.x;
    const int b   = bt / TFRAMES;
    const int t   = bt - b * TFRAMES;
    const int tid = threadIdx.x;
    const float a1 = a1p[0];
    const float a2 = a2p[0];

    if (tid < 16) {
        x1[tid * 485 + 0]   = 0.0f;
        x1[tid * 485 + 1]   = 0.0f;
        x1[tid * 485 + 483] = 0.0f;
        x1[tid * 485 + 484] = 0.0f;
    }

    for (int f = tid; f < NFREQ; f += 256) {
        float v[4][9];
        #pragma unroll
        for (int d = 0; d < 4; ++d) {
            #pragma unroll
            for (int k = 0; k < 9; ++k) {
                int ff = f + k - 4;
                v[d][k] = (ff >= 0 && ff < NFREQ)
                    ? feat[(((size_t)b * 4 + d) * NFREQ + ff) * TFRAMES + t]
                    : 0.0f;
            }
        }
        #pragma unroll
        for (int c = 0; c < 16; ++c) {
            float a = b1[c];
            #pragma unroll
            for (int d = 0; d < 4; ++d)
                #pragma unroll
                for (int k = 0; k < 9; ++k)
                    a = fmaf(v[d][k], w1[(c * 4 + d) * 9 + k], a);
            a = (a >= 0.0f) ? a : a1 * a;
            x1[c * 485 + f + 2] = a;
        }
    }
    __syncthreads();

    for (int f = tid; f < NFREQ; f += 256) {
        float u[16][5];
        #pragma unroll
        for (int ci = 0; ci < 16; ++ci)
            #pragma unroll
            for (int k = 0; k < 5; ++k)
                u[ci][k] = x1[ci * 485 + f + k];

        union { __bf16 h[16]; uint4 q[2]; } res;
        #pragma unroll
        for (int co = 0; co < 16; ++co) {
            float a = b2[co];
            #pragma unroll
            for (int ci = 0; ci < 16; ++ci)
                #pragma unroll
                for (int k = 0; k < 5; ++k)
                    a = fmaf(u[ci][k], w2[(co * 16 + ci) * 5 + k], a);
            a = (a >= 0.0f) ? a : a2 * a;
            res.h[co] = (__bf16)a;
        }
        __bf16* dst = xout + ((size_t)t * NBINS + (size_t)b * NFREQ + f) * 16;
        *(uint4*)(dst + 0) = res.q[0];
        *(uint4*)(dst + 8) = res.q[1];
    }
}

// =====================================================================
// Stage 2: GRU scan + fused FC/sigmoid.
// One wave per block, 16 bins per block, 481 blocks.
// Per step: r/z gates via chained WMMA (x then h accumulating into the
// same C), n gate split so r*hn is possible. Zero C operand (shared),
// biases folded into gate VALU as per-lane scalars -> low VGPR pressure.
// =====================================================================
__global__ __launch_bounds__(32) void fbd_gru_wmma_kernel(
    const __bf16* __restrict__ xbf,   // (T, NBINS, 16) bf16
    const float* __restrict__ h0,     // (1, NBINS, 32)
    const float* __restrict__ w_ih,   // (96, 16)
    const float* __restrict__ w_hh,   // (96, 32)
    const float* __restrict__ b_ih,   // (96)
    const float* __restrict__ b_hh,   // (96)
    const float* __restrict__ fc_w,   // (1, 32)
    const float* __restrict__ fc_b,   // (1)
    float* __restrict__ out_prob,     // (NBINS, T) == (B, F, T)
    float* __restrict__ out_h)        // (NBINS, 32)
{
    __shared__ __bf16 hsh[16 * 32];   // h tile, [bin][k], bf16

    const int lane = threadIdx.x;
    const int lo   = lane & 15;       // A: M / B,C: column N
    const int g    = lane >> 4;       // A: K-group / C: M-group
    const int n0   = blockIdx.x * 16;

    // ---- pack B matrices (constant over the whole scan) ----
    // B (32x16 bf16, 8 VGPRs): lane half g, e0..7 = K 8g..8g+7,
    //                          e8..15 = K 16+8g..16+8g+7 (N=lane%16).
    v16bf Bhh[6], Bih[6];
    #pragma unroll
    for (int j = 0; j < 6; ++j) {
        const float* sh = w_hh + (size_t)(16 * j + lo) * 32 + g * 8;
        const float* si = w_ih + (size_t)(16 * j + lo) * 16 + g * 8;
        v16bf bh, bi;
        #pragma unroll
        for (int e = 0; e < 16; ++e) {
            bh[e] = (__bf16)sh[e + (e & 8)];
            float wv = si[e & 7];             // K>=16 rows of x are zero-padded
            bi[e] = (e & 8) ? (__bf16)0.0f : (__bf16)wv;
        }
        Bhh[j] = bh;
        Bih[j] = bi;
    }

    // per-lane scalar biases (column = 16j + lo)
    const float bs0   = b_ih[lo]      + b_hh[lo];        // r gate, cols 0-15
    const float bs1   = b_ih[16 + lo] + b_hh[16 + lo];   // r gate, cols 16-31
    const float bs2   = b_ih[32 + lo] + b_hh[32 + lo];   // z gate, cols 0-15
    const float bs3   = b_ih[48 + lo] + b_hh[48 + lo];   // z gate, cols 16-31
    const float bin0  = b_ih[64 + lo];
    const float bhn0  = b_hh[64 + lo];
    const float bin1  = b_ih[80 + lo];
    const float bhn1  = b_hh[80 + lo];
    const float fcwA  = fc_w[lo];
    const float fcwB  = fc_w[16 + lo];
    const float fcb   = fc_b[0];

    // ---- initial hidden state (C layout: row M = 8g+r, col = 16j+lo) ----
    float hC0[8], hC1[8];
    #pragma unroll
    for (int r = 0; r < 8; ++r) {
        hC0[r] = h0[(size_t)(n0 + 8 * g + r) * 32 + lo];
        hC1[r] = h0[(size_t)(n0 + 8 * g + r) * 32 + 16 + lo];
        hsh[(8 * g + r) * 32 + lo]      = (__bf16)hC0[r];
        hsh[(8 * g + r) * 32 + 16 + lo] = (__bf16)hC1[r];
    }
    // single wave: same-wave DS ops are in-order, no barrier needed

    v8bf z8;
    #pragma unroll
    for (int e = 0; e < 8; ++e) z8[e] = (__bf16)0.0f;
    const v8f zc = {0.f, 0.f, 0.f, 0.f, 0.f, 0.f, 0.f, 0.f};

    // strength-reduced pointers
    const size_t xstep = (size_t)NBINS * 16;                  // bf16 elems / frame
    const __bf16* xp   = xbf + ((size_t)n0 + lo) * 16 + 8 * g;
    float*        opb  = out_prob + (size_t)(n0 + 8 * g) * TFRAMES;

    v8bf xv = *(const v8bf*)xp;                               // x_0 preloaded

    for (int t = 0; t < TFRAMES; ++t) {
        // A-matrix of h (16x32 bf16): lane m=lo, e0..7 = K 8g..8g+7,
        //                                        e8..15 = K 16+8g..16+8g+7
        v8bf hl = *(const v8bf*)(&hsh[lo * 32 + 8 * g]);
        v8bf hh = *(const v8bf*)(&hsh[lo * 32 + 16 + 8 * g]);
        v16bf hA = __builtin_shufflevector(hl, hh,
            0, 1, 2, 3, 4, 5, 6, 7, 8, 9, 10, 11, 12, 13, 14, 15);
        v16bf xA = __builtin_shufflevector(xv, z8,
            0, 1, 2, 3, 4, 5, 6, 7, 8, 9, 10, 11, 12, 13, 14, 15);

        // r/z gates: x-proj and h-proj chained into one accumulator per tile
        v8f acc0 = __builtin_amdgcn_wmma_f32_16x16x32_bf16(false, xA, false, Bih[0], (short)0, zc, false, false);
        v8f acc1 = __builtin_amdgcn_wmma_f32_16x16x32_bf16(false, xA, false, Bih[1], (short)0, zc, false, false);
        v8f acc2 = __builtin_amdgcn_wmma_f32_16x16x32_bf16(false, xA, false, Bih[2], (short)0, zc, false, false);
        v8f acc3 = __builtin_amdgcn_wmma_f32_16x16x32_bf16(false, xA, false, Bih[3], (short)0, zc, false, false);
        acc0 = __builtin_amdgcn_wmma_f32_16x16x32_bf16(false, hA, false, Bhh[0], (short)0, acc0, false, false);
        acc1 = __builtin_amdgcn_wmma_f32_16x16x32_bf16(false, hA, false, Bhh[1], (short)0, acc1, false, false);
        acc2 = __builtin_amdgcn_wmma_f32_16x16x32_bf16(false, hA, false, Bhh[2], (short)0, acc2, false, false);
        acc3 = __builtin_amdgcn_wmma_f32_16x16x32_bf16(false, hA, false, Bhh[3], (short)0, acc3, false, false);
        // n gate: keep x and h parts separate (r multiplies only the h part)
        v8f xgn0 = __builtin_amdgcn_wmma_f32_16x16x32_bf16(false, xA, false, Bih[4], (short)0, zc, false, false);
        v8f ghn0 = __builtin_amdgcn_wmma_f32_16x16x32_bf16(false, hA, false, Bhh[4], (short)0, zc, false, false);
        v8f xgn1 = __builtin_amdgcn_wmma_f32_16x16x32_bf16(false, xA, false, Bih[5], (short)0, zc, false, false);
        v8f ghn1 = __builtin_amdgcn_wmma_f32_16x16x32_bf16(false, hA, false, Bhh[5], (short)0, zc, false, false);

        // prefetch next frame's x while gate math runs
        const __bf16* xpn = xp + ((t < TFRAMES - 1) ? xstep : 0);
        v8bf xv_next = *(const v8bf*)xpn;
        xp = xpn;

        float p[8];
        #pragma unroll
        for (int r = 0; r < 8; ++r) {
            float rr0 = sigm_f(acc0[r] + bs0);
            float zz0 = sigm_f(acc2[r] + bs2);
            float nn0 = tanh_f(xgn0[r] + bin0 + rr0 * (ghn0[r] + bhn0));
            float hn0 = (1.0f - zz0) * nn0 + zz0 * hC0[r];

            float rr1 = sigm_f(acc1[r] + bs1);
            float zz1 = sigm_f(acc3[r] + bs3);
            float nn1 = tanh_f(xgn1[r] + bin1 + rr1 * (ghn1[r] + bhn1));
            float hn1 = (1.0f - zz1) * nn1 + zz1 * hC1[r];

            hC0[r] = hn0;
            hC1[r] = hn1;
            p[r]   = hn0 * fcwA + hn1 * fcwB;

            hsh[(8 * g + r) * 32 + lo]      = (__bf16)hn0;
            hsh[(8 * g + r) * 32 + 16 + lo] = (__bf16)hn1;
        }

        // FC reduction across the 16 columns (lanes sharing the same g)
        #pragma unroll
        for (int r = 0; r < 8; ++r) {
            float s = p[r];
            s += __shfl_xor(s, 1);
            s += __shfl_xor(s, 2);
            s += __shfl_xor(s, 4);
            s += __shfl_xor(s, 8);
            p[r] = s;
        }
        if (lo == 0) {
            #pragma unroll
            for (int r = 0; r < 8; ++r)
                opb[(size_t)r * TFRAMES] = sigm_f(p[r] + fcb);
        }
        opb += 1;
        xv = xv_next;
    }

    // final hidden state
    #pragma unroll
    for (int r = 0; r < 8; ++r) {
        out_h[(size_t)(n0 + 8 * g + r) * 32 + lo]      = hC0[r];
        out_h[(size_t)(n0 + 8 * g + r) * 32 + 16 + lo] = hC1[r];
    }
}

// =====================================================================
extern "C" void kernel_launch(void* const* d_in, const int* in_sizes, int n_in,
                              void* d_out, int out_size, void* d_ws, size_t ws_size,
                              hipStream_t stream) {
    (void)in_sizes; (void)n_in; (void)out_size; (void)ws_size;

    const float* feat = (const float*)d_in[0];   // features
    const float* h0   = (const float*)d_in[1];   // h
    const float* w1   = (const float*)d_in[2];   // conv1_w
    const float* b1   = (const float*)d_in[3];   // conv1_b
    const float* a1   = (const float*)d_in[4];   // prelu1
    const float* w2   = (const float*)d_in[5];   // conv2_w
    const float* b2   = (const float*)d_in[6];   // conv2_b
    const float* a2   = (const float*)d_in[7];   // prelu2
    const float* wih  = (const float*)d_in[8];   // w_ih
    const float* whh  = (const float*)d_in[9];   // w_hh
    const float* bih  = (const float*)d_in[10];  // b_ih
    const float* bhh  = (const float*)d_in[11];  // b_hh
    const float* fcw  = (const float*)d_in[12];  // fc_w
    const float* fcb  = (const float*)d_in[13];  // fc_b

    float* out_prob = (float*)d_out;                              // (B,F,T)
    float* out_h    = out_prob + (size_t)NBINS * TFRAMES;         // (1,N,32)
    __bf16* xbf     = (__bf16*)d_ws;                              // (T,N,16) bf16

    fbd_conv_kernel<<<BB * TFRAMES, 256, 0, stream>>>(
        feat, w1, b1, a1, w2, b2, a2, xbf);

    fbd_gru_wmma_kernel<<<NBINS / 16, 32, 0, stream>>>(
        xbf, h0, wih, whh, bih, bhh, fcw, fcb, out_prob, out_h);
}
